// SparseMoE_64707977282229
// MI455X (gfx1250) — matrix operand order repaired and checked by reading, not compiled, not run on hardware
//
#include <hip/hip_runtime.h>
#include <stdint.h>

typedef __attribute__((ext_vector_type(16))) __bf16 v16bf;
typedef __attribute__((ext_vector_type(8)))  __bf16 v8bf;
typedef __attribute__((ext_vector_type(8)))  float  v8f;
typedef __attribute__((ext_vector_type(4)))  uint32_t u32x4;
typedef __attribute__((ext_vector_type(8)))  uint32_t u32x8;

union Frag16 { v16bf v; v8bf h[2]; };
union BPack { __bf16 b[2]; uint32_t u; };

#define WMMA_BF16(a, b, c) \
    __builtin_amdgcn_wmma_f32_16x16x32_bf16(false, (a), false, (b), (short)0, (c), false, false)

// ---------------------------------------------------------------- init
__global__ __launch_bounds__(256) void init_kernel(int* idList, float* ones,
                                                   int* counts, int N) {
    int i = blockIdx.x * 256 + threadIdx.x;
    if (i < N) { idList[i] = i; ones[i] = 1.0f; }
    if (i < 8) counts[i] = 0;
    if (i == 8) counts[8] = N;   // "count" for the shared expert
}

// ---------------------------------------------------------------- x -> bf16
__global__ __launch_bounds__(256) void hconvert_kernel(const float* __restrict__ x,
                                                       __bf16* __restrict__ hbf,
                                                       int total) {
    int i = (blockIdx.x * 256 + threadIdx.x) * 4;
    if (i >= total) return;
    float4 v = *(const float4*)(x + i);
    hbf[i + 0] = (__bf16)v.x;
    hbf[i + 1] = (__bf16)v.y;
    hbf[i + 2] = (__bf16)v.z;
    hbf[i + 3] = (__bf16)v.w;
}

// ---------------------------------------------------------------- router
__global__ __launch_bounds__(256) void router_kernel(
    const float* __restrict__ x, const float* __restrict__ Wgate,
    float* __restrict__ scores, int* __restrict__ counts,
    int* __restrict__ tokList, float* __restrict__ wList, int N, int D) {
    int lane = threadIdx.x & 31;
    int wid  = threadIdx.x >> 5;
    int token = blockIdx.x * 8 + wid;
    if (token >= N) return;

    float acc[8] = {0.f, 0.f, 0.f, 0.f, 0.f, 0.f, 0.f, 0.f};
    for (int d = lane; d < D; d += 32) {
        float xv = x[(size_t)token * D + d];
#pragma unroll
        for (int e = 0; e < 8; ++e) acc[e] += xv * Wgate[d * 8 + e];
    }
#pragma unroll
    for (int e = 0; e < 8; ++e) {
#pragma unroll
        for (int off = 16; off > 0; off >>= 1)
            acc[e] += __shfl_xor(acc[e], off, 32);
    }
    if (lane == 0) {
        float mx = acc[0];
#pragma unroll
        for (int e = 1; e < 8; ++e) mx = fmaxf(mx, acc[e]);
        float p[8], sum = 0.f;
#pragma unroll
        for (int e = 0; e < 8; ++e) { p[e] = __expf(acc[e] - mx); sum += p[e]; }
#pragma unroll
        for (int e = 0; e < 8; ++e) {
            p[e] /= sum;
            scores[(size_t)token * 8 + e] = p[e];
        }
        int i0 = 0; float s0 = p[0];
#pragma unroll
        for (int e = 1; e < 8; ++e) if (p[e] > s0) { s0 = p[e]; i0 = e; }
        int i1 = (i0 == 0) ? 1 : 0; float s1 = -1.f;
#pragma unroll
        for (int e = 0; e < 8; ++e)
            if (e != i0 && p[e] > s1) { s1 = p[e]; i1 = e; }
        float denom = s0 + s1 + 1e-9f;
        float w0 = s0 / denom, w1 = s1 / denom;
        int slot0 = atomicAdd(&counts[i0], 1);
        tokList[i0 * N + slot0] = token; wList[i0 * N + slot0] = w0;
        int slot1 = atomicAdd(&counts[i1], 1);
        tokList[i1 * N + slot1] = token; wList[i1 * N + slot1] = w1;
    }
}

// ---------------------------------------------------------------- aux loss
__global__ __launch_bounds__(256) void aux_kernel(const float* __restrict__ scores,
                                                  const int* __restrict__ counts,
                                                  float* __restrict__ auxOut, int N) {
    __shared__ float sh[8 * 256];
    int tid = threadIdx.x;
    float p[8] = {0.f, 0.f, 0.f, 0.f, 0.f, 0.f, 0.f, 0.f};
    for (int n = tid; n < N; n += 256) {
#pragma unroll
        for (int e = 0; e < 8; ++e) p[e] += scores[(size_t)n * 8 + e];
    }
#pragma unroll
    for (int e = 0; e < 8; ++e) sh[e * 256 + tid] = p[e];
    __syncthreads();
    for (int s = 128; s > 0; s >>= 1) {
        if (tid < s) {
#pragma unroll
            for (int e = 0; e < 8; ++e) sh[e * 256 + tid] += sh[e * 256 + tid + s];
        }
        __syncthreads();
    }
    if (tid == 0) {
        float aux = 0.f;
#pragma unroll
        for (int e = 0; e < 8; ++e) {
            float P = sh[e * 256] / (float)N;
            float f = (float)counts[e] / (2.0f * N + 1e-9f);
            aux += f * P;
        }
        auxOut[0] = 0.01f * 8.0f * aux;
    }
}

// ---------------------------------------------------------------- gate+up fused GEMM
// t[slot, i] = silu(h[tok]@Wg)[i] * (h[tok]@Wu)[i], tile 128(rows) x 64(cols), K=32 steps
__global__ __launch_bounds__(256) void gateup_kernel(
    const __bf16* __restrict__ hbf, const float* __restrict__ Wg,
    const float* __restrict__ Wu, const int* __restrict__ tokList,
    const int* __restrict__ countPtr, __bf16* __restrict__ tOut,
    int D, int I) {
    const int count = *countPtr;
    const int rowBase = blockIdx.y * 128;
    if (rowBase >= count) return;
    const int colBase = blockIdx.x * 64;

    __shared__ __align__(16) __bf16 Asm[128][40];
    __shared__ __align__(16) __bf16 Bgs[64][40];
    __shared__ __align__(16) __bf16 Bus[64][40];

    const int tid  = threadIdx.x;
    const int lane = tid & 31;
    const int wid  = tid >> 5;
    const int hs   = lane >> 4;
    const int l15  = lane & 15;
    const int waveM = wid & 3;   // 4 row groups of 32
    const int waveN = wid >> 2;  // 2 col groups of 32

    // A-load role: one 32-wide bf16 row chunk pair per thread (gathered rows)
    const int arow = tid >> 1;
    const int aq   = (tid & 1) * 2;           // quarters (8 bf16 each)
    int grow = rowBase + arow; if (grow >= count) grow = count - 1;
    const int atok = tokList[grow];
    // B-load role: 4 cols x 2 adjacent K rows -> packed b32 LDS stores
    const int nq = (tid & 15) * 4;            // 0..60
    const int kp = (tid >> 4) * 2;            // 0..30

    v8f accG[2][2], accU[2][2];
    const v8f vzero = {0.f, 0.f, 0.f, 0.f, 0.f, 0.f, 0.f, 0.f};
#pragma unroll
    for (int mt = 0; mt < 2; ++mt)
#pragma unroll
        for (int nt = 0; nt < 2; ++nt) { accG[mt][nt] = vzero; accU[mt][nt] = vzero; }

    for (int kk = 0; kk < D; kk += 32) {
        const uint4* gA = (const uint4*)(hbf + (size_t)atok * D + kk);
        uint4 a0 = gA[aq];
        uint4 a1 = gA[aq + 1];
        const float* gG0 = Wg + (size_t)(kk + kp) * I + colBase + nq;
        float4 g0 = *(const float4*)gG0;
        float4 g1 = *(const float4*)(gG0 + I);
        const float* gU0 = Wu + (size_t)(kk + kp) * I + colBase + nq;
        float4 u0 = *(const float4*)gU0;
        float4 u1 = *(const float4*)(gU0 + I);

        __syncthreads();
        *(uint4*)&Asm[arow][aq * 8]     = a0;
        *(uint4*)&Asm[arow][aq * 8 + 8] = a1;
        {
            BPack p;
            p.b[0] = (__bf16)g0.x; p.b[1] = (__bf16)g1.x; *(uint32_t*)&Bgs[nq + 0][kp] = p.u;
            p.b[0] = (__bf16)g0.y; p.b[1] = (__bf16)g1.y; *(uint32_t*)&Bgs[nq + 1][kp] = p.u;
            p.b[0] = (__bf16)g0.z; p.b[1] = (__bf16)g1.z; *(uint32_t*)&Bgs[nq + 2][kp] = p.u;
            p.b[0] = (__bf16)g0.w; p.b[1] = (__bf16)g1.w; *(uint32_t*)&Bgs[nq + 3][kp] = p.u;
            p.b[0] = (__bf16)u0.x; p.b[1] = (__bf16)u1.x; *(uint32_t*)&Bus[nq + 0][kp] = p.u;
            p.b[0] = (__bf16)u0.y; p.b[1] = (__bf16)u1.y; *(uint32_t*)&Bus[nq + 1][kp] = p.u;
            p.b[0] = (__bf16)u0.z; p.b[1] = (__bf16)u1.z; *(uint32_t*)&Bus[nq + 2][kp] = p.u;
            p.b[0] = (__bf16)u0.w; p.b[1] = (__bf16)u1.w; *(uint32_t*)&Bus[nq + 3][kp] = p.u;
        }
        __syncthreads();

        Frag16 aF[2];
#pragma unroll
        for (int mt = 0; mt < 2; ++mt) {
            int m = waveM * 32 + mt * 16 + l15;
            aF[mt].h[0] = *(const v8bf*)&Asm[m][hs * 8];
            aF[mt].h[1] = *(const v8bf*)&Asm[m][16 + hs * 8];
        }
        Frag16 bG[2], bU[2];
#pragma unroll
        for (int nt = 0; nt < 2; ++nt) {
            int n = waveN * 32 + nt * 16 + l15;
            bG[nt].h[0] = *(const v8bf*)&Bgs[n][hs * 8];
            bG[nt].h[1] = *(const v8bf*)&Bgs[n][16 + hs * 8];
            bU[nt].h[0] = *(const v8bf*)&Bus[n][hs * 8];
            bU[nt].h[1] = *(const v8bf*)&Bus[n][16 + hs * 8];
        }
#pragma unroll
        for (int mt = 0; mt < 2; ++mt)
#pragma unroll
            for (int nt = 0; nt < 2; ++nt) {
                accG[mt][nt] = WMMA_BF16(aF[mt].v, bG[nt].v, accG[mt][nt]);
                accU[mt][nt] = WMMA_BF16(aF[mt].v, bU[nt].v, accU[mt][nt]);
            }
    }

#pragma unroll
    for (int mt = 0; mt < 2; ++mt)
#pragma unroll
        for (int nt = 0; nt < 2; ++nt) {
            int gr0 = rowBase + waveM * 32 + mt * 16 + hs * 8;
            int gc  = colBase + waveN * 32 + nt * 16 + l15;
#pragma unroll
            for (int r = 0; r < 8; ++r) {
                float g = accG[mt][nt][r];
                float u = accU[mt][nt][r];
                float sig = __builtin_amdgcn_rcpf(1.0f + __expf(-g));
                float val = g * sig * u;
                tOut[(size_t)(gr0 + r) * I + gc] = (__bf16)val;
            }
        }
}

// ---------------------------------------------------------------- down GEMM + combine
// A-tile (t buffer, contiguous slot-major bf16) is fetched by the Tensor Data
// Mover: 2D tile 128 x 32 bf16, LDS padding 16B per 64B row -> 80B LDS stride.
__global__ __launch_bounds__(256) void down_kernel(
    const __bf16* __restrict__ tIn, const float* __restrict__ Wd,
    const int* __restrict__ tokList, const float* __restrict__ wList,
    const int* __restrict__ countPtr, float* __restrict__ out,
    int I, int D, int accumulate) {
    const int count = *countPtr;
    const int rowBase = blockIdx.y * 128;
    if (rowBase >= count) return;
    const int colBase = blockIdx.x * 64;

    __shared__ __align__(16) __bf16 Asm[128][40];
    __shared__ __align__(16) __bf16 Bds[64][40];

    const int tid  = threadIdx.x;
    const int lane = tid & 31;
    const int wid  = tid >> 5;
    const int hs   = lane >> 4;
    const int l15  = lane & 15;
    const int waveM = wid & 3;
    const int waveN = wid >> 2;

    const int nq = (tid & 15) * 4;
    const int kp = (tid >> 4) * 2;

    v8f acc[2][2];
    const v8f vzero = {0.f, 0.f, 0.f, 0.f, 0.f, 0.f, 0.f, 0.f};
#pragma unroll
    for (int mt = 0; mt < 2; ++mt)
#pragma unroll
        for (int nt = 0; nt < 2; ++nt) acc[mt][nt] = vzero;

    const uint32_t ldsA = (uint32_t)(uintptr_t)(&Asm[0][0]);

    for (int kk = 0; kk < I; kk += 32) {
        const float* gD0 = Wd + (size_t)(kk + kp) * D + colBase + nq;
        float4 d0 = *(const float4*)gD0;
        float4 d1 = *(const float4*)(gD0 + D);

        __syncthreads();   // previous compute finished reading LDS

        if (wid == 0) {
            // ---- Tensor DMA: 128 x 32 bf16 tile of tIn -> Asm (padded rows)
            uint64_t ga = (uint64_t)(uintptr_t)(tIn + (size_t)rowBase * I + kk);
            u32x4 g0v;
            g0v.x = 1u;                                   // count=1, no gather
            g0v.y = ldsA;                                 // lds_addr
            g0v.z = (uint32_t)ga;                         // global_addr[31:0]
            g0v.w = (uint32_t)(ga >> 32) | (2u << 30);    // global_addr[56:32] | type=2
            u32x8 g1v;
            g1v[0] = (1u << 16)        // data_size = 2 bytes
                   | (1u << 20)        // pad_enable
                   | (3u << 22)        // pad_interval: every 16 DWORDs (64B)
                   | (3u << 25);       // pad_amount: 4 DWORDs (16B)
            g1v[1] = ((uint32_t)I & 0xFFFFu) << 16;                 // tensor_dim0 lo16
            g1v[2] = (((uint32_t)I >> 16) & 0xFFFFu);               // tensor_dim0 hi16 | dim1 lo16(=0)
            g1v[3] = 0x10u | (32u << 16);                           // tensor_dim1 hi16 (1M) | tile_dim0=32
            g1v[4] = 128u;                                          // tile_dim1 = 128 rows
            g1v[5] = (uint32_t)I;                                   // tensor_dim0_stride lo32
            g1v[6] = 0u;
            g1v[7] = 0u;
            asm volatile("tensor_load_to_lds %0, %1" :: "s"(g0v), "s"(g1v) : "memory");
        }

        {
            BPack p;
            p.b[0] = (__bf16)d0.x; p.b[1] = (__bf16)d1.x; *(uint32_t*)&Bds[nq + 0][kp] = p.u;
            p.b[0] = (__bf16)d0.y; p.b[1] = (__bf16)d1.y; *(uint32_t*)&Bds[nq + 1][kp] = p.u;
            p.b[0] = (__bf16)d0.z; p.b[1] = (__bf16)d1.z; *(uint32_t*)&Bds[nq + 2][kp] = p.u;
            p.b[0] = (__bf16)d0.w; p.b[1] = (__bf16)d1.w; *(uint32_t*)&Bds[nq + 3][kp] = p.u;
        }

        if (wid == 0) __builtin_amdgcn_s_wait_tensorcnt(0);
        __syncthreads();   // A (TDM) + B (ds stores) visible to all waves

        Frag16 aF[2];
#pragma unroll
        for (int mt = 0; mt < 2; ++mt) {
            int m = waveM * 32 + mt * 16 + l15;
            aF[mt].h[0] = *(const v8bf*)&Asm[m][hs * 8];
            aF[mt].h[1] = *(const v8bf*)&Asm[m][16 + hs * 8];
        }
        Frag16 bF[2];
#pragma unroll
        for (int nt = 0; nt < 2; ++nt) {
            int n = waveN * 32 + nt * 16 + l15;
            bF[nt].h[0] = *(const v8bf*)&Bds[n][hs * 8];
            bF[nt].h[1] = *(const v8bf*)&Bds[n][16 + hs * 8];
        }
#pragma unroll
        for (int mt = 0; mt < 2; ++mt)
#pragma unroll
            for (int nt = 0; nt < 2; ++nt)
                acc[mt][nt] = WMMA_BF16(aF[mt].v, bF[nt].v, acc[mt][nt]);
    }

#pragma unroll
    for (int mt = 0; mt < 2; ++mt)
#pragma unroll
        for (int nt = 0; nt < 2; ++nt) {
            int slot0 = rowBase + waveM * 32 + mt * 16 + hs * 8;
            int gc    = colBase + waveN * 32 + nt * 16 + l15;
#pragma unroll
            for (int r = 0; r < 8; ++r) {
                int slot = slot0 + r;
                if (slot < count) {
                    int tok = tokList[slot];
                    float w = wList[slot];
                    float val = acc[mt][nt][r] * w;
                    size_t idx = (size_t)tok * D + gc;
                    if (accumulate) out[idx] += val;
                    else            out[idx]  = val;
                }
            }
        }
}

// ---------------------------------------------------------------- launch
extern "C" void kernel_launch(void* const* d_in, const int* in_sizes, int n_in,
                              void* d_out, int out_size, void* d_ws, size_t ws_size,
                              hipStream_t stream) {
    (void)in_sizes; (void)n_in; (void)out_size; (void)ws_size;
    const float* x       = (const float*)d_in[0];
    const float* Wg      = (const float*)d_in[1];
    const float* W_gate  = (const float*)d_in[2];
    const float* W_up    = (const float*)d_in[3];
    const float* W_down  = (const float*)d_in[4];
    const float* Ws_gate = (const float*)d_in[5];
    const float* Ws_up   = (const float*)d_in[6];
    const float* Ws_down = (const float*)d_in[7];
    float* out = (float*)d_out;

    const int Bv = 2, Sv = 2048, D = 1024, I = 2048, E = 8;
    const int N = Bv * Sv;   // 4096

    char* w = (char*)d_ws;
    size_t off = 0;
    auto alloc = [&](size_t bytes) -> void* {
        void* p = w + off;
        off = (off + bytes + 255) & ~(size_t)255;
        return p;
    };
    __bf16* hbf    = (__bf16*)alloc((size_t)N * D * 2);
    __bf16* tbf    = (__bf16*)alloc((size_t)N * I * 2);
    float*  scores = (float*) alloc((size_t)N * E * 4);
    int*    tokL   = (int*)   alloc((size_t)E * N * 4);
    float*  wL     = (float*) alloc((size_t)E * N * 4);
    int*    idL    = (int*)   alloc((size_t)N * 4);
    float*  ones   = (float*) alloc((size_t)N * 4);
    int*    counts = (int*)   alloc(64);   // [0..7]=expert counts, [8]=N

    init_kernel<<<(N + 255) / 256, 256, 0, stream>>>(idL, ones, counts, N);
    hconvert_kernel<<<(N * D / 4 + 255) / 256, 256, 0, stream>>>(x, hbf, N * D);
    router_kernel<<<N / 8, 256, 0, stream>>>(x, Wg, scores, counts, tokL, wL, N, D);
    aux_kernel<<<1, 256, 0, stream>>>(scores, counts, out + (size_t)N * D, N);

    dim3 gu(I / 64, N / 128);   // 32 x 32
    dim3 gd(D / 64, N / 128);   // 16 x 32

    // shared expert: assign pass (initializes out fully)
    gateup_kernel<<<gu, 256, 0, stream>>>(hbf, Ws_gate, Ws_up, idL, counts + 8, tbf, D, I);
    down_kernel<<<gd, 256, 0, stream>>>(tbf, Ws_down, idL, ones, counts + 8, out, I, D, 0);

    // routed experts: gathered, accumulate
    for (int e = 0; e < E; ++e) {
        gateup_kernel<<<gu, 256, 0, stream>>>(hbf,
            W_gate + (size_t)e * D * I, W_up + (size_t)e * D * I,
            tokL + e * N, counts + e, tbf, D, I);
        down_kernel<<<gd, 256, 0, stream>>>(tbf,
            W_down + (size_t)e * I * D,
            tokL + e * N, wL + e * N, counts + e, out, I, D, 1);
    }
}
